// GroupQueryAttention_14276471291955
// MI455X (gfx1250) — matrix operand order repaired and checked by reading, not compiled, not run on hardware
//
#include <hip/hip_runtime.h>

typedef unsigned short ushort_t;
typedef __attribute__((ext_vector_type(16))) __bf16 v16bf;
typedef __attribute__((ext_vector_type(8)))  float  v8f;

#define B_   2
#define S_   2048
#define E_   2048
#define NH_  32
#define NKV_ 8
#define HD_  64
#define GS_  4

// ---------------- bf16 bit helpers ----------------
__device__ inline ushort_t f2us(float f) {
    union { float f; unsigned u; } v; v.f = f;
    unsigned r = v.u + 0x7FFFu + ((v.u >> 16) & 1u);   // round-to-nearest-even
    return (ushort_t)(r >> 16);
}

// ---------------- WMMA fragment loaders (k-contiguous per lane, not the ISA 7.12.2 table k order) ----
// All addresses are 16-byte aligned by construction; uint4 accesses force
// ds_load_b128 / global_load_b128.

// 32 contiguous bytes -> one v16bf fragment (per-lane k-contiguous layout).
__device__ inline v16bf frag_from_2xb128(const ushort_t* p) {
    uint4 a = *(const uint4*)p;
    uint4 b = *(const uint4*)(p + 8);
    v16bf r;
    __builtin_memcpy(&r, &a, 16);
    __builtin_memcpy((char*)&r + 16, &b, 16);
    return r;
}

// A 16x32 row-major [m][k]: lane m=lane&15, h=lane>>4.
// Per-lane elements: k in [16h, 16h+16) (contiguous)  -> two b128 loads.
__device__ inline v16bf load_a_frag(const ushort_t* base, int ld, int lane) {
    int m = lane & 15, h = lane >> 4;
    uint4 a = *(const uint4*)(base + m * ld + 16 * h);
    uint4 b = *(const uint4*)(base + m * ld + 16 * h + 8);
    v16bf r;
    __builtin_memcpy(&r, &a, 16);
    __builtin_memcpy((char*)&r + 16, &b, 16);
    return r;
}

// B 32x16 from TRANSPOSED storage [n][k] (ld = k-stride in elements):
// lane n=lane&15, h=lane>>4; element e -> k = e + 16h  (contiguous per lane).
__device__ inline v16bf load_b_fragT(const ushort_t* baseT, int ld, int lane) {
    int n = lane & 15, h = lane >> 4;
    return frag_from_2xb128(baseT + n * ld + 16 * h);
}

__device__ inline v8f wmma_bf16(v16bf a, v16bf b, v8f c) {
    return __builtin_amdgcn_wmma_f32_16x16x32_bf16(false, a, false, b, (short)0, c, false, false);
}

// ---------------- f32 -> bf16 convert (vectorized) ----------------
__global__ void cvt_f32_bf16(const float* __restrict__ in, ushort_t* __restrict__ out, int n) {
    int i = (blockIdx.x * blockDim.x + threadIdx.x) * 4;
    if (i >= n) return;
    float4 f = *(const float4*)(in + i);
    union { ushort_t u[4]; uint2 v; } o;
    o.u[0] = f2us(f.x); o.u[1] = f2us(f.y); o.u[2] = f2us(f.z); o.u[3] = f2us(f.w);
    *(uint2*)(out + i) = o.v;
}

// ---------------- bf16 GEMM: C[M,N](f32) = A[M,K] @ B[K,N], tiles 128x64x32 ----
#define TM 128
#define TN 64
#define TK 32
__global__ __launch_bounds__(256) void gemm_bf16(const ushort_t* __restrict__ A,
                                                 const ushort_t* __restrict__ B,
                                                 float* __restrict__ C,
                                                 int M, int N, int K) {
    __shared__ ushort_t As[TM][TK];     // row-major [m][k]
    __shared__ ushort_t Bs[TN][TK];     // TRANSPOSED [n][k] -> b128 fragment reads
    int tid  = threadIdx.x;
    int wave = tid >> 5, lane = tid & 31;
    int bm = blockIdx.x * TM;
    int bn = blockIdx.y * TN;
    v8f acc[4] = {};

    // hoisted staging addresses (advance by fixed strides per k-step)
    int ar = tid >> 1, acb = (tid & 1) * 16;      // A: row, col-chunk
    int bk = tid >> 3, bnb = (tid & 7) * 8;       // B: k-row, n-chunk
    const ushort_t* aptr = A + (size_t)(bm + ar) * K + acb;
    const ushort_t* bptr = B + (size_t)bk * N + bn + bnb;

    for (int k0 = 0; k0 < K; k0 += TK) {
        { // A tile: 2x global b128 -> 2x ds b128 (issue both loads first)
            uint4 a0 = ((const uint4*)aptr)[0];
            uint4 a1 = ((const uint4*)aptr)[1];
            uint4* dst = (uint4*)&As[ar][acb];
            dst[0] = a0;
            dst[1] = a1;
        }
        { // B tile: one coalesced global b128 (8 contiguous n at fixed k),
          // scatter transposed into Bs[n][k] with 8 independent b16 stores
            union { ushort_t u[8]; uint4 v; } tmp;
            tmp.v = *(const uint4*)bptr;
#pragma unroll
            for (int j = 0; j < 8; ++j)
                Bs[bnb + j][bk] = tmp.u[j];
        }
        aptr += TK;
        bptr += (size_t)TK * N;
        if (k0 + TK < K) {                          // gfx1250 global_prefetch path
            __builtin_prefetch(aptr, 0, 1);
            __builtin_prefetch(bptr, 0, 1);
        }
        __syncthreads();
        v16bf a = load_a_frag(&As[wave * 16][0], TK, lane);
#pragma unroll
        for (int j = 0; j < 4; ++j) {
            v16bf b = load_b_fragT(&Bs[16 * j][0], TK, lane);
            acc[j] = wmma_bf16(a, b, acc[j]);
        }
        __syncthreads();
    }
    int h = lane >> 4, n = lane & 15;
#pragma unroll
    for (int j = 0; j < 4; ++j)
#pragma unroll
        for (int i = 0; i < 8; ++i) {
            int row = bm + wave * 16 + i + 8 * h;
            int col = bn + 16 * j + n;
            C[(size_t)row * N + col] = acc[j][i];
        }
}

// ---------------- fused RMSNorm + RoPE, one wave per (b,s,head) row ----------
__global__ __launch_bounds__(256) void norm_rope(const float* __restrict__ X,
                                                 const float* __restrict__ w,
                                                 ushort_t* __restrict__ outB,
                                                 float* __restrict__ outF,
                                                 int nh, float scale, int total) {
    int wid  = blockIdx.x * (blockDim.x >> 5) + (threadIdx.x >> 5);
    int lane = threadIdx.x & 31;
    if (wid >= total) return;
    int h = wid % nh;
    int s = (wid / nh) % S_;
    int b = wid / (nh * S_);

    const float* row = X + ((size_t)b * S_ + s) * (nh * HD_) + h * HD_;
    float x1 = row[lane], x2 = row[lane + 32];
    float ss = x1 * x1 + x2 * x2;
#pragma unroll
    for (int off = 16; off > 0; off >>= 1) ss += __shfl_xor(ss, off, 32);
    float inv = rsqrtf(ss * (1.0f / 64.0f) + 1e-6f);
    x1 = x1 * inv * w[lane];
    x2 = x2 * inv * w[lane + 32];

    float freq = (float)s * __expf(-((float)lane / 32.0f) * 9.2103403719761836f); // 10000^(-2l/64)
    float c = __cosf(freq), sn = __sinf(freq);
    float o1 = (x1 * c - x2 * sn) * scale;
    float o2 = (x2 * c + x1 * sn) * scale;

    size_t dst = (((size_t)b * nh + h) * S_ + s) * HD_;
    outB[dst + lane]      = f2us(o1);
    outB[dst + lane + 32] = f2us(o2);
    if (outF) { outF[dst + lane] = o1; outF[dst + lane + 32] = o2; }
}

// ---------------- V: (b,s,kv*64+d) f32 -> bf16 (b,kv,s,d) + f32 cache out ----
__global__ void v_pack(const float* __restrict__ Vf, ushort_t* __restrict__ Vb,
                       float* __restrict__ outV, int total) {
    int i = blockIdx.x * blockDim.x + threadIdx.x;
    if (i >= total) return;
    int d  = i & 63;
    int s  = (i >> 6) % S_;
    int kv = ((i >> 6) / S_) % NKV_;
    int b  = i / (HD_ * S_ * NKV_);
    float v = Vf[((size_t)b * S_ + s) * (NKV_ * HD_) + kv * HD_ + d];
    Vb[i]   = f2us(v);
    outV[i] = v;
}

// ---------------- flash attention: one wave per (b, head, 16-query tile) ----
__global__ __launch_bounds__(32) void attn_fwd(const ushort_t* __restrict__ Qb,
                                               const ushort_t* __restrict__ Kb,
                                               const ushort_t* __restrict__ Vb,
                                               ushort_t* __restrict__ Ctx) {
    __shared__ ushort_t Ps[16 * 32];      // P row-major [m][k]
    __shared__ ushort_t Vs_t[64 * 32];    // V TRANSPOSED [d][key] -> b128 frag reads
    int lane = threadIdx.x;
    int q0 = blockIdx.x * 16;
    int h  = blockIdx.y;
    int b  = blockIdx.z;
    int kv = h / GS_;
    int hh = lane >> 4, nn = lane & 15;

    const ushort_t* qbase = Qb + (((size_t)b * NH_ + h) * S_ + q0) * HD_;
    v16bf qa0 = load_a_frag(qbase,      HD_, lane);  // d 0..31 (1/sqrt(HD) pre-folded)
    v16bf qa1 = load_a_frag(qbase + 32, HD_, lane);  // d 32..63

    const ushort_t* kbase = Kb + (((size_t)b * NKV_ + kv) * S_) * HD_;
    const ushort_t* vbase = Vb + (((size_t)b * NKV_ + kv) * S_) * HD_;

    float mrun[8], lrun[8];
    v8f acc[4] = {};
#pragma unroll
    for (int i = 0; i < 8; ++i) { mrun[i] = -3.0e38f; lrun[i] = 0.0f; }

    int ntiles = (q0 + 16 + 31) / 32;
    for (int t = 0; t < ntiles; ++t) {
        int t0 = t * 32;
        if (t + 1 < ntiles) {
            __builtin_prefetch(kbase + (size_t)(t0 + 32 + lane) * HD_, 0, 1);
            __builtin_prefetch(vbase + (size_t)(t0 + 32 + lane) * HD_, 0, 1);
        }
        // ---- scores S(16x32) = Q(16x64) @ K^T(64x32); K rows are k-contiguous
        v8f sc[2] = {};
#pragma unroll
        for (int g = 0; g < 2; ++g) {
            const ushort_t* kb = kbase + (size_t)(t0 + 16 * g) * HD_;
            sc[g] = wmma_bf16(qa0, load_b_fragT(kb,      HD_, lane), sc[g]);
            sc[g] = wmma_bf16(qa1, load_b_fragT(kb + 32, HD_, lane), sc[g]);
        }
        // ---- stage V tile transposed: lane reads its key row (8x b128),
        //      scatters to Vs_t[d][key] (conflict-free b16 stores)
        {
            union { ushort_t u[64]; uint4 v[8]; } rowv;
            const uint4* vsrc = (const uint4*)(vbase + (size_t)(t0 + lane) * HD_);
#pragma unroll
            for (int i = 0; i < 8; ++i) rowv.v[i] = vsrc[i];
#pragma unroll
            for (int d = 0; d < 64; ++d) Vs_t[d * 32 + lane] = rowv.u[d];
        }
        // ---- causal mask + online softmax (row reductions over 16-lane halves)
#pragma unroll
        for (int i = 0; i < 8; ++i) {
            int r = i + 8 * hh, q = q0 + r;
#pragma unroll
            for (int g = 0; g < 2; ++g) {
                int key = t0 + 16 * g + nn;
                if (key > q) sc[g][i] = -1.0e30f;
            }
            float m = fmaxf(sc[0][i], sc[1][i]);
#pragma unroll
            for (int off = 8; off > 0; off >>= 1) m = fmaxf(m, __shfl_xor(m, off, 32));
            float mnew  = fmaxf(mrun[i], m);
            float alpha = __expf(mrun[i] - mnew);
            mrun[i] = mnew;
            float p0 = __expf(sc[0][i] - mnew);
            float p1 = __expf(sc[1][i] - mnew);
            float rs = p0 + p1;
#pragma unroll
            for (int off = 8; off > 0; off >>= 1) rs += __shfl_xor(rs, off, 32);
            lrun[i] = lrun[i] * alpha + rs;
#pragma unroll
            for (int j = 0; j < 4; ++j) acc[j][i] *= alpha;
            Ps[r * 32 + nn]      = f2us(p0);
            Ps[r * 32 + 16 + nn] = f2us(p1);
        }
        __syncthreads();
        // ---- acc(16x64) += P(16x32) @ V(32x64)
        v16bf pa = load_a_frag(Ps, 32, lane);
#pragma unroll
        for (int j = 0; j < 4; ++j) {
            v16bf vb = load_b_fragT(&Vs_t[16 * j * 32], 32, lane);
            acc[j] = wmma_bf16(pa, vb, acc[j]);
        }
        __syncthreads();
    }
    // ---- normalize, write ctx bf16 (b, s, h*64+d)
#pragma unroll
    for (int j = 0; j < 4; ++j)
#pragma unroll
        for (int i = 0; i < 8; ++i) {
            int r = i + 8 * hh;
            float o = acc[j][i] / lrun[i];
            Ctx[((size_t)b * S_ + q0 + r) * (NH_ * HD_) + h * HD_ + 16 * j + nn] = f2us(o);
        }
}

// ---------------- host-side orchestration ----------------
extern "C" void kernel_launch(void* const* d_in, const int* in_sizes, int n_in,
                              void* d_out, int out_size, void* d_ws, size_t ws_size,
                              hipStream_t stream) {
    const float* x   = (const float*)d_in[0];
    // d_in[1] position_ids, d_in[2] attn_mask: causal/positions derived analytically
    const float* Wq  = (const float*)d_in[3];
    const float* Wk  = (const float*)d_in[4];
    const float* Wv  = (const float*)d_in[5];
    const float* Wo  = (const float*)d_in[6];
    const float* qnw = (const float*)d_in[7];
    const float* knw = (const float*)d_in[8];
    float* out = (float*)d_out;

    char* ws = (char*)d_ws;
    // workspace layout (bytes)
    const size_t XB   = 0;                           // x bf16        16 MB
    const size_t WQB  = XB   + (size_t)B_*S_*E_*2;   // Wq bf16        8 MB
    const size_t WKB  = WQB  + (size_t)E_*NH_*HD_*2; // Wk bf16        2 MB
    const size_t WVB  = WKB  + (size_t)E_*NKV_*HD_*2;
    const size_t WOB  = WVB  + (size_t)E_*NKV_*HD_*2;
    const size_t QF   = WOB  + (size_t)NH_*HD_*E_*2; // Q f32         32 MB
    const size_t KF   = QF   + (size_t)B_*S_*NH_*HD_*4;
    const size_t VF   = KF   + (size_t)B_*S_*NKV_*HD_*4;
    const size_t QBUF = VF   + (size_t)B_*S_*NKV_*HD_*4; // Q bf16 (b,h,s,d)
    const size_t KBUF = QBUF + (size_t)B_*NH_*S_*HD_*2;
    const size_t VBUF = KBUF + (size_t)B_*NKV_*S_*HD_*2;
    const size_t CTX  = VBUF + (size_t)B_*NKV_*S_*HD_*2; // ctx bf16 (b,s,2048)

    ushort_t* xb  = (ushort_t*)(ws + XB);
    ushort_t* wqb = (ushort_t*)(ws + WQB);
    ushort_t* wkb = (ushort_t*)(ws + WKB);
    ushort_t* wvb = (ushort_t*)(ws + WVB);
    ushort_t* wob = (ushort_t*)(ws + WOB);
    float*    qf  = (float*)(ws + QF);
    float*    kf  = (float*)(ws + KF);
    float*    vf  = (float*)(ws + VF);
    ushort_t* qbuf = (ushort_t*)(ws + QBUF);
    ushort_t* kbuf = (ushort_t*)(ws + KBUF);
    ushort_t* vbuf = (ushort_t*)(ws + VBUF);
    ushort_t* ctx  = (ushort_t*)(ws + CTX);

    const int M = B_ * S_;          // 4096
    // 1) bf16 converts (4 elems/thread)
    auto cvt = [&](const float* src, ushort_t* dst, int n) {
        cvt_f32_bf16<<<(n / 4 + 255) / 256, 256, 0, stream>>>(src, dst, n);
    };
    cvt(x,  xb,  B_ * S_ * E_);
    cvt(Wq, wqb, E_ * NH_ * HD_);
    cvt(Wk, wkb, E_ * NKV_ * HD_);
    cvt(Wv, wvb, E_ * NKV_ * HD_);
    cvt(Wo, wob, NH_ * HD_ * E_);

    // 2) projections (WMMA GEMMs)
    gemm_bf16<<<dim3(M / TM, (NH_ * HD_) / TN), 256, 0, stream>>>(xb, wqb, qf, M, NH_ * HD_, E_);
    gemm_bf16<<<dim3(M / TM, (NKV_ * HD_) / TN), 256, 0, stream>>>(xb, wkb, kf, M, NKV_ * HD_, E_);
    gemm_bf16<<<dim3(M / TM, (NKV_ * HD_) / TN), 256, 0, stream>>>(xb, wvb, vf, M, NKV_ * HD_, E_);

    // 3) RMSNorm + RoPE (Q gets 1/sqrt(HD) folded in); K also to f32 cache output
    const size_t outK = (size_t)B_ * S_ * E_;                   // 8,388,608
    const size_t outV = outK + (size_t)B_ * NKV_ * S_ * HD_;    // 10,485,760
    {
        int totalQ = B_ * S_ * NH_;   // waves
        norm_rope<<<totalQ / 8, 256, 0, stream>>>(qf, qnw, qbuf, nullptr, NH_, 0.125f, totalQ);
        int totalK = B_ * S_ * NKV_;
        norm_rope<<<totalK / 8, 256, 0, stream>>>(kf, knw, kbuf, out + outK, NKV_, 1.0f, totalK);
        int totalV = B_ * NKV_ * S_ * HD_;
        v_pack<<<(totalV + 255) / 256, 256, 0, stream>>>(vf, vbuf, out + outV, totalV);
    }

    // 4) flash attention (WMMA QK^T and PV)
    attn_fwd<<<dim3(S_ / 16, NH_, B_), 32, 0, stream>>>(qbuf, kbuf, vbuf, ctx);

    // 5) output projection straight into d_out
    gemm_bf16<<<dim3(M / TM, E_ / TN), 256, 0, stream>>>(ctx, wob, out, M, E_, NH_ * HD_);
}
